// MRNN_72756745994367
// MI455X (gfx1250) — matrix operand order repaired and checked by reading, not compiled, not run on hardware
//
#include <hip/hip_runtime.h>
#include <hip/hip_bf16.h>

// ---------------- feature probes (compile-safe) ----------------
#ifndef HAVE_TDM
#if defined(__has_builtin)
#  if __has_builtin(__builtin_amdgcn_tensor_load_to_lds)
#    define HAVE_TDM 1
#  else
#    define HAVE_TDM 0
#  endif
#else
#  define HAVE_TDM 0
#endif
#endif

#if defined(__has_include)
#  if __has_include(<hip/amd_detail/amd_gfx1250_TDM.h>)
#    define TDM_SIX_ARG 1
#  else
#    define TDM_SIX_ARG 0
#  endif
#else
#  define TDM_SIX_ARG 0
#endif

// ---- CDNA5 WMMA / vector types ----
typedef __attribute__((ext_vector_type(16))) __bf16 v16bf;
typedef __attribute__((ext_vector_type(8)))  __bf16 bf16x8;
typedef __attribute__((ext_vector_type(8)))  float  v8f;
typedef __attribute__((ext_vector_type(2)))  float  f32x2;
typedef unsigned int u32x4 __attribute__((ext_vector_type(4)));
typedef int          i32x8 __attribute__((ext_vector_type(8)));
typedef int          i32x4 __attribute__((ext_vector_type(4)));

#define B_DIM 256
#define C_DIM 64
#define T_DIM 2048
#define T_OUT 2046   // T - 2
#define T_PAD 2048   // padded rows in h1 workspace

// ============================================================
// Stage 1: fused fwd+bwd scalar RNN scan + w1 projection.
// Thread = one sequence (b,c); both recurrences advance in step-
// index order (fwd from the front, bwd from the back), so
// h1[t] = relu(w1_0*hf + w1_1*hb + b1) falls out in a single pass.
// h1 layout: [b][t(pad 2048)][c] bf16 (wave writes 64 contig bytes/t).
// ============================================================

template<int CNT>
__device__ __forceinline__ void rnn_chunk(
    const float* __restrict__ xs, int t0,
    float wf0, float wf1, float wf2, float BF,
    float wb0, float wb1, float wb2, float BB,
    float uf, float ub, float w10, float w11, float B1,
    float& hf, float& hb, __bf16* __restrict__ hrow)
{
    float af[CNT], ab[CNT];
    const float* pf = xs + (size_t)t0 * 3;
    const float* pb = xs + (size_t)(T_DIM - CNT - t0) * 3;

#pragma unroll
    for (int i = 0; i < CNT; ++i) {
        const float* p = pf + i * 3;
        af[i] = fmaf(p[2], wf2, fmaf(p[1], wf1, fmaf(p[0], wf0, BF)));
    }
#pragma unroll
    for (int i = 0; i < CNT; ++i) {
        const float* q = pb + (CNT - 1 - i) * 3;   // row r <-> step CNT-1-r
        ab[i] = fmaf(q[2], wb2, fmaf(q[1], wb1, fmaf(q[0], wb0, BB)));
    }
#pragma unroll
    for (int i = 0; i < CNT; ++i) {
        hf = fmaxf(fmaf(uf, hf, af[i]), 0.f);
        hb = fmaxf(fmaf(ub, hb, ab[i]), 0.f);
        float v = fmaxf(fmaf(w10, hf, fmaf(w11, hb, B1)), 0.f);
        hrow[(size_t)(t0 + i) * C_DIM] = (__bf16)v;
    }
}

__global__ __launch_bounds__(256)
void mrnn_scan_kernel(const float* __restrict__ x,
                      const float* __restrict__ Wf, const float* __restrict__ Uf,
                      const float* __restrict__ bf,
                      const float* __restrict__ Wb, const float* __restrict__ Ub,
                      const float* __restrict__ bb,
                      const float* __restrict__ w1, const float* __restrict__ b1,
                      __bf16* __restrict__ h1)
{
    int s = blockIdx.x * 256 + threadIdx.x;      // 0..16383
    int b = s >> 6;
    int c = s & 63;

    const float* xs = x + (size_t)s * T_DIM * 3;

    float wf0 = Wf[0], wf1 = Wf[1], wf2 = Wf[2], BF = bf[0];
    float wb0 = Wb[0], wb1 = Wb[1], wb2 = Wb[2], BB = bb[0];
    float uf = Uf[0], ub = Ub[0];
    float w10 = w1[0], w11 = w1[1], B1 = b1[0];

    __bf16* hrow = h1 + (size_t)b * T_PAD * C_DIM + c;
    hrow[(size_t)2046 * C_DIM] = (__bf16)0.f;    // zero pad rows
    hrow[(size_t)2047 * C_DIM] = (__bf16)0.f;

    float hf = 0.f, hb = 0.f;
    for (int t0 = 0; t0 < 2040; t0 += 8) {
        if (t0 + 16 < 2040) {
            __builtin_prefetch(xs + (size_t)(t0 + 16) * 3, 0, 0);
            __builtin_prefetch(xs + (size_t)(T_DIM - 8 - (t0 + 16)) * 3, 0, 0);
        }
        rnn_chunk<8>(xs, t0, wf0, wf1, wf2, BF, wb0, wb1, wb2, BB,
                     uf, ub, w10, w11, B1, hf, hb, hrow);
    }
    rnn_chunk<6>(xs, 2040, wf0, wf1, wf2, BF, wb0, wb1, wb2, BB,
                 uf, ub, w10, w11, B1, hf, hb, hrow);
}

// ============================================================
// Stage 2: channel-mixing GEMM via v_wmma_f32_16x16x32_bf16.
//   out[b,c,t] = relu( sum_k h1[b][t][k] * w2[k][c] + b2[c] )
// A tiles [16 t x 64 k] staged LDS via the Tensor Data Mover
// (double-buffered, one 2KB DMA per tile per wave), B = w2
// fragments resident in VGPRs, D = [16 t x 16 c] f32.
// ============================================================

#if HAVE_TDM
__device__ __forceinline__ void tdm_load_tile(unsigned long long gaddr,
                                              unsigned int lds_off)
{
    // D# group0: count=1 | lds_addr | global_addr[56:0] | type=2 ("image")
    u32x4 g0;
    g0.x = 1u;
    g0.y = lds_off;
    g0.z = (unsigned int)gaddr;
    g0.w = (unsigned int)(gaddr >> 32) | (2u << 30);

    // D# group1: 2D tensor, data_size=2B, tensor 64 x 2048, tile 64 x 16,
    // dim0 stride = 64 elements.
    i32x8 g1;
    g1[0] = 0x00010000;            // data_size = 1 (2 bytes); mask/flags = 0
    g1[1] = (int)(64u << 16);      // tensor_dim0 = 64      (bits 79:48)
    g1[2] = (int)(2048u << 16);    // tensor_dim1 = 2048    (bits 111:80)
    g1[3] = (int)(64u << 16);      // tile_dim0 = 64        (bits 127:112)
    g1[4] = 16;                    // tile_dim1 = 16        (bits 143:128)
    g1[5] = 64;                    // tensor_dim0_stride    (bits 207:160)
    g1[6] = 0;
    g1[7] = 0;

    i32x4 z4 = {0, 0, 0, 0};
#if TDM_SIX_ARG
    i32x8 z8 = {0, 0, 0, 0, 0, 0, 0, 0};
    __builtin_amdgcn_tensor_load_to_lds(g0, g1, z4, z4, z8, 0);
#else
    __builtin_amdgcn_tensor_load_to_lds(g0, g1, z4, z4, 0);
#endif
}

template<int N>
__device__ __forceinline__ void wait_tensorcnt()
{
#if defined(__has_builtin)
#if __has_builtin(__builtin_amdgcn_s_wait_tensorcnt)
    __builtin_amdgcn_s_wait_tensorcnt(N);
#else
    asm volatile("s_wait_tensorcnt %0" :: "i"(N) : "memory");
#endif
#else
    asm volatile("s_wait_tensorcnt %0" :: "i"(N) : "memory");
#endif
    asm volatile("" ::: "memory");   // keep LDS reads below the wait
}
#endif // HAVE_TDM

__device__ __forceinline__ void epilogue_store(const v8f& d, float bs,
                                               float* __restrict__ op, int nv)
{
    f32x2 v01 = { fmaxf(d[0] + bs, 0.f), fmaxf(d[1] + bs, 0.f) };
    f32x2 v23 = { fmaxf(d[2] + bs, 0.f), fmaxf(d[3] + bs, 0.f) };
    f32x2 v45 = { fmaxf(d[4] + bs, 0.f), fmaxf(d[5] + bs, 0.f) };
    *(f32x2*)(op + 0) = v01;
    *(f32x2*)(op + 2) = v23;
    *(f32x2*)(op + 4) = v45;
    if (nv == 4) {                 // only lanes 16-31 of tile t0=2032 skip this
        f32x2 v67 = { fmaxf(d[6] + bs, 0.f), fmaxf(d[7] + bs, 0.f) };
        *(f32x2*)(op + 6) = v67;
    }
}

__device__ __forceinline__ void gemm_tile(const __bf16* __restrict__ arow,
                                          const v16bf Bf[4][2],
                                          const float bias[4],
                                          int b, int t0, int n, int half,
                                          float* __restrict__ out)
{
    // A fragment: row M = n, K runs [base..base+7] -> slots 0-7,
    // [base+16..base+23] -> slots 8-15, base = ks*32 + half*8.
    v16bf A[2];
#pragma unroll
    for (int ks = 0; ks < 2; ++ks) {
        bf16x8 lo = *(const bf16x8*)(arow + ks * 32);
        bf16x8 hi = *(const bf16x8*)(arow + ks * 32 + 16);
#pragma unroll
        for (int i = 0; i < 8; ++i) { A[ks][i] = lo[i]; A[ks][i + 8] = hi[i]; }
    }

    // Batch all 8 WMMAs: 4 independent accumulators hide the
    // WMMA->VALU hazard slots before the epilogue.
    v8f d[4];
#pragma unroll
    for (int ct = 0; ct < 4; ++ct) {
        v8f z = {0.f, 0.f, 0.f, 0.f, 0.f, 0.f, 0.f, 0.f};
        d[ct] = __builtin_amdgcn_wmma_f32_16x16x32_bf16(
                    false, A[0], false, Bf[ct][0], (short)0, z, false, false);
    }
#pragma unroll
    for (int ct = 0; ct < 4; ++ct) {
        d[ct] = __builtin_amdgcn_wmma_f32_16x16x32_bf16(
                    false, A[1], false, Bf[ct][1], (short)0, d[ct], false, false);
    }

    int trow = t0 + half * 8;
    int nv   = (t0 == 2032 && half) ? 3 : 4;   // t >= 2046 is padding
#pragma unroll
    for (int ct = 0; ct < 4; ++ct) {
        int cc = ct * 16 + n;
        float* op = out + (size_t)(b * 64 + cc) * T_OUT + trow;
        epilogue_store(d[ct], bias[ct], op, nv);
    }
}

__global__ __launch_bounds__(256)
void mrnn_gemm_kernel(const __bf16* __restrict__ h1,
                      const float* __restrict__ w2,
                      const float* __restrict__ b2,
                      float* __restrict__ out)
{
    int b    = blockIdx.x;
    int tid  = threadIdx.x;
    int wave = tid >> 5;       // 8 waves
    int lane = tid & 31;
    int n    = lane & 15;      // tile column (c)
    int half = lane >> 4;

    // --- B fragments: w2 as [K=32 x N=16] bf16 tiles, 4 c-tiles x 2 k-slabs
    v16bf Bf[4][2];
    float bias[4];
#pragma unroll
    for (int ct = 0; ct < 4; ++ct) {
        bias[ct] = b2[ct * 16 + n];
#pragma unroll
        for (int ks = 0; ks < 2; ++ks) {
#pragma unroll
            for (int i = 0; i < 16; ++i) {
                int k = ks * 32 + half * 8 + (i < 8 ? i : i + 8);
                Bf[ct][ks][i] = (__bf16)w2[k * 64 + ct * 16 + n];
            }
        }
    }

#if HAVE_TDM
    // Per-wave double-buffered LDS staging of [16 t x 64 c] bf16 tiles (2KB).
    __shared__ __attribute__((aligned(16))) __bf16 smem[8][2][16 * 64];

    unsigned long long gbase =
        (unsigned long long)(uintptr_t)(h1 + (size_t)b * T_PAD * C_DIM);
    unsigned int lds_off[2] = {
        (unsigned int)(uintptr_t)&smem[wave][0][0],
        (unsigned int)(uintptr_t)&smem[wave][1][0]
    };

    // prime the pipeline: tile i=0 (tt = wave)
    tdm_load_tile(gbase + (unsigned long long)(wave * 16) * C_DIM * 2, lds_off[0]);

    for (int i = 0; i < 16; ++i) {               // tt = wave + 8*i
        int t0  = (wave + 8 * i) * 16;
        int par = i & 1;
        if (i + 1 < 16) {
            int ttn = wave + 8 * (i + 1);
            tdm_load_tile(gbase + (unsigned long long)(ttn * 16) * C_DIM * 2,
                          lds_off[par ^ 1]);
            wait_tensorcnt<1>();                 // current buffer is ready
        } else {
            wait_tensorcnt<0>();
        }
        const __bf16* arow = &smem[wave][par][0] + n * C_DIM + half * 8;
        gemm_tile(arow, Bf, bias, b, t0, n, half, out);
    }
#else
    // Fallback: direct 16B global loads from h1 (row-major A tiles).
    const __bf16* h1b = h1 + (size_t)b * T_PAD * C_DIM;
    for (int tt = wave; tt < 128; tt += 8) {
        int t0 = tt * 16;
        const __bf16* arow = h1b + (size_t)(t0 + n) * C_DIM + half * 8;
        gemm_tile(arow, Bf, bias, b, t0, n, half, out);
    }
#endif
}

// ============================================================
// Launch
// ============================================================
extern "C" void kernel_launch(void* const* d_in, const int* in_sizes, int n_in,
                              void* d_out, int out_size, void* d_ws, size_t ws_size,
                              hipStream_t stream)
{
    const float* x  = (const float*)d_in[0];
    const float* Wf = (const float*)d_in[1];
    const float* Uf = (const float*)d_in[2];
    const float* bf = (const float*)d_in[3];
    const float* Wb = (const float*)d_in[4];
    const float* Ub = (const float*)d_in[5];
    const float* bb = (const float*)d_in[6];
    const float* w1 = (const float*)d_in[7];
    const float* b1 = (const float*)d_in[8];
    const float* w2 = (const float*)d_in[9];
    const float* b2 = (const float*)d_in[10];

    __bf16* h1  = (__bf16*)d_ws;   // 256*2048*64*2 = 64 MiB
    float*  out = (float*)d_out;

    hipLaunchKernelGGL(mrnn_scan_kernel, dim3(16384 / 256), dim3(256), 0, stream,
                       x, Wf, Uf, bf, Wb, Ub, bb, w1, b1, h1);
    hipLaunchKernelGGL(mrnn_gemm_kernel, dim3(B_DIM), dim3(256), 0, stream,
                       h1, w2, b2, out);
}
